// CrossSparseAggrNet_v2_41197326303535
// MI455X (gfx1250) — matrix operand order repaired and checked by reading, not compiled, not run on hardware
//
#include <hip/hip_runtime.h>
#include <hip/hip_bf16.h>

// ---------------- problem constants ----------------
#define CDIM   512
#define LV     196
#define BV     64
#define BT     32
#define NWORD  24
#define NKEEP  98          // ceil(196*0.5)
#define NTAIL  98          // 196-98
#define KEEPED 49          // int(196*0.25)
#define HIDDEN 102         // int(512*0.2)
#define LAMBDA 4.0f
#define LNEPS  1e-5f

// padded tile dims for WMMA (16x16x32 bf16)
#define MP   112           // 98 -> 112   (7 M tiles)
#define HPAD 112           // 102 -> 112  (7 N tiles)
#define PPAD 64            // 49 -> 64    (4 M tiles)
#define SSN  520           // selnorm row stride (floats)

// LDS strides (in dwords) chosen for bank-conflict avoidance
#define XSTRW 260          // xnb row stride   (520 bf16): 260%64=4 -> A-frag rows hit distinct banks
#define TOKW  516          // tokpk row stride: 8*516%64=32 -> B-frag half-waves 32 banks apart
#define WABW  68           // wab row stride   (136 bf16): 68%64=4

// W1 fragment-major: dword idx = (((kk*7+nt)*32)+lane)*8+g ; 16*7*32*8 = 28672 dwords
#define W1FM_DWORDS (16 * 7 * 32 * 8)

typedef __attribute__((ext_vector_type(16))) __bf16 v16bf;
typedef __attribute__((ext_vector_type(2)))  __bf16 bf16x2;
typedef __attribute__((ext_vector_type(8)))  float  v8f;

// ---------------- k_main dynamic LDS layout (bytes) ----------------
#define XNB_OFF   0                 // 112*260*4 = 116480 bf16 xn  [aliased: wtmp f32, selnorm f32 50x520]
#define TOK_OFF   116480            // 64*516*4  = 132096 packed tokens [aliased early: W1 frag-major 114688]
#define WAB_OFF   248576            // 64*68*4   = 17408  packed softmax weights
#define H_OFF     265984            // 98*112*4  = 43904  f32 h
#define KIX_OFF   309888            // 98 ints (padded)
#define WSIM_OFF  310400            // 24 floats (padded)
#define SMEM_TOTAL 310528

__device__ inline float lane_sum(float x) {
#pragma unroll
  for (int m = 16; m >= 1; m >>= 1) x += __shfl_xor(x, m, 32);
  return x;
}
__device__ inline float lane_max(float x) {
#pragma unroll
  for (int m = 16; m >= 1; m >>= 1) x = fmaxf(x, __shfl_xor(x, m, 32));
  return x;
}
__device__ inline unsigned pack_bf16(float a, float b) {
  bf16x2 p; p[0] = (__bf16)a; p[1] = (__bf16)b;
  return __builtin_bit_cast(unsigned, p);
}

union Frag  { v16bf v; unsigned u[8]; };
union FragQ { v16bf v; uint4 q[2]; };

// A row-major bf16 [M][2*lduw]; dword view, K-pairs contiguous. 8x b32 loads.
__device__ inline v16bf load_a_frag_pk(const unsigned* A, int lduw, int m0, int k0, int lane) {
  int row  = m0 + (lane & 15);
  int kadd = (lane & 16) ? 8 : 0;            // dword units (16 bf16 = 8 dwords)
  const unsigned* rp = A + (size_t)row * lduw + (k0 >> 1);
  Frag f;
#pragma unroll
  for (int g = 0; g < 8; ++g) {
    int kp = kadd + g;
    f.u[g] = rp[kp];
  }
  return f.v;
}
// B pair-packed: dword (k>>1)*lduw + n holds bf16 pair (k, k+1) of column n. 8x b32 loads.
__device__ inline v16bf load_b_frag_pk(const unsigned* Bpk, int lduw, int k0, int n0, int lane) {
  int col = n0 + (lane & 15);
  int kb  = (lane & 16) ? 8 : 0;             // 16 K-elems = 8 dword rows
  const unsigned* cp = Bpk + (size_t)((k0 >> 1) + kb) * lduw + col;
  Frag f;
#pragma unroll
  for (int g = 0; g < 8; ++g) f.u[g] = cp[(size_t)g * lduw];
  return f.v;
}
// B fragment-major: lane's 8 dwords contiguous -> 2x ds_load_b128, zero marshalling.
__device__ inline v16bf load_b_frag_fm(const unsigned* Bfm, int kk, int nt, int lane) {
  const uint4* p = (const uint4*)(Bfm + ((((size_t)kk * 7 + nt) * 32) + lane) * 8);
  FragQ f;
  f.q[0] = p[0];
  f.q[1] = p[1];
  return f.v;
}

// =====================================================================
// Kernel 1a: per-image preprocessing (img_glo, per-token inv-norm, self_attn)
// =====================================================================
__global__ void __launch_bounds__(256) k_prep_img(const float* __restrict__ img,
                                                  float* __restrict__ img_glo,
                                                  float* __restrict__ inv_img,
                                                  float* __restrict__ self_attn) {
  __shared__ float glo[CDIM];
  __shared__ float red[8];
  int v = blockIdx.x, tid = threadIdx.x, wid = tid >> 5, lane = tid & 31;
  const float* base = img + (size_t)v * LV * CDIM;

  float m0 = 0.f, m1 = 0.f;
  for (int l = 0; l < LV; ++l) {
    m0 += base[l * CDIM + tid];
    m1 += base[l * CDIM + tid + 256];
  }
  m0 *= (1.f / LV); m1 *= (1.f / LV);
  float p = lane_sum(m0 * m0 + m1 * m1);
  if (lane == 0) red[wid] = p;
  __syncthreads();
  float tot = 0.f;
#pragma unroll
  for (int i = 0; i < 8; ++i) tot += red[i];
  float inv = 1.f / fmaxf(sqrtf(tot), 1e-12f);
  float g0 = m0 * inv, g1 = m1 * inv;
  glo[tid] = g0; glo[tid + 256] = g1;
  img_glo[(size_t)v * CDIM + tid] = g0;
  img_glo[(size_t)v * CDIM + tid + 256] = g1;
  __syncthreads();

  for (int l = wid; l < LV; l += 8) {
    const float* tp = base + l * CDIM;
    float ss = 0.f, dot = 0.f;
#pragma unroll
    for (int i = 0; i < 16; ++i) {
      int c = lane + 32 * i;
      float x = tp[c];
      ss += x * x; dot += glo[c] * x;
    }
    ss = lane_sum(ss); dot = lane_sum(dot);
    if (lane == 0) {
      float il = 1.f / fmaxf(sqrtf(ss), 1e-12f);
      inv_img[(size_t)v * LV + l] = il;
      self_attn[(size_t)v * LV + l] = dot * il;
    }
  }
}

// =====================================================================
// Kernel 1b: per-caption preprocessing (cap_glo, cap_norm)
// =====================================================================
__global__ void __launch_bounds__(256) k_prep_cap(const float* __restrict__ cap,
                                                  float* __restrict__ cap_glo,
                                                  float* __restrict__ cap_nrm) {
  __shared__ float red[8];
  int t = blockIdx.x, tid = threadIdx.x, wid = tid >> 5, lane = tid & 31;
  const float* base = cap + (size_t)t * NWORD * CDIM;
  float m0 = 0.f, m1 = 0.f;
  for (int w = 0; w < NWORD; ++w) {
    m0 += base[w * CDIM + tid];
    m1 += base[w * CDIM + tid + 256];
  }
  m0 *= (1.f / NWORD); m1 *= (1.f / NWORD);
  float p = lane_sum(m0 * m0 + m1 * m1);
  if (lane == 0) red[wid] = p;
  __syncthreads();
  float tot = 0.f;
#pragma unroll
  for (int i = 0; i < 8; ++i) tot += red[i];
  float inv = 1.f / fmaxf(sqrtf(tot), 1e-12f);
  cap_glo[(size_t)t * CDIM + tid] = m0 * inv;
  cap_glo[(size_t)t * CDIM + tid + 256] = m1 * inv;

  for (int w = wid; w < NWORD; w += 8) {
    const float* wp = base + w * CDIM;
    float vals[16]; float ss = 0.f;
#pragma unroll
    for (int i = 0; i < 16; ++i) {
      int c = lane + 32 * i;
      float x = wp[c]; vals[i] = x; ss += x * x;
    }
    ss = lane_sum(ss);
    float iw = 1.f / fmaxf(sqrtf(ss), 1e-12f);
#pragma unroll
    for (int i = 0; i < 16; ++i) {
      int c = lane + 32 * i;
      cap_nrm[((size_t)t * NWORD + w) * CDIM + c] = vals[i] * iw;
    }
  }
}

// =====================================================================
// Kernel 1c: W1 (512x102 f32) -> fragment-major pair-packed bf16 dwords.
// Output dword idx = (((kk*7+nt)*32)+lane)*8+g holds bf16 pair
// (W1[k][col], W1[k+1][col]) with k = 2*(kk*16 + kb + g), kb = (lane&16)?8:0,
// col = nt*16 + (lane&15)  -- exactly one lane-fragment dword of the B operand.
// =====================================================================
__global__ void k_w1cvt(const float* __restrict__ W1, unsigned* __restrict__ W1fm) {
  int idx = blockIdx.x * blockDim.x + threadIdx.x;
  if (idx >= W1FM_DWORDS) return;
  int g    = idx & 7;
  int lane = (idx >> 3) & 31;
  int r    = idx >> 8;          // kk*7 + nt
  int nt   = r % 7;
  int kk   = r / 7;
  int col  = nt * 16 + (lane & 15);
  int kb   = (lane & 16) ? 8 : 0;
  int k0   = 2 * (kk * 16 + kb + g);
  float a = 0.f, b = 0.f;
  if (col < HIDDEN) {
    a = W1[(size_t)k0 * HIDDEN + col];
    b = W1[(size_t)(k0 + 1) * HIDDEN + col];
  }
  W1fm[idx] = pack_bf16(a, b);
}

// =====================================================================
// Kernel 2: score[t][v][l] = self_attn[v][l] + cap_glo[t].img_norm[v,l]
// =====================================================================
__global__ void __launch_bounds__(256) k_score(const float* __restrict__ img,
                                               const float* __restrict__ inv_img,
                                               const float* __restrict__ self_attn,
                                               const float* __restrict__ cap_glo,
                                               float* __restrict__ score) {
  __shared__ float tok[CDIM];
  int l = blockIdx.x, v = blockIdx.y;
  int tid = threadIdx.x, wid = tid >> 5, lane = tid & 31;
  float inv = inv_img[(size_t)v * LV + l];
  const float* tp = img + ((size_t)v * LV + l) * CDIM;
  tok[tid] = tp[tid] * inv;
  tok[tid + 256] = tp[tid + 256] * inv;
  __syncthreads();
  float sa = self_attn[(size_t)v * LV + l];
  int t0 = wid * 4;
  float acc[4] = {0.f, 0.f, 0.f, 0.f};
  for (int i = 0; i < 16; ++i) {
    int c = lane + 32 * i;
    float x = tok[c];
#pragma unroll
    for (int q = 0; q < 4; ++q) acc[q] += x * cap_glo[(size_t)(t0 + q) * CDIM + c];
  }
#pragma unroll
  for (int q = 0; q < 4; ++q) {
    float d = lane_sum(acc[q]);
    if (lane == 0) score[((size_t)(t0 + q) * BV + v) * LV + l] = sa + d;
  }
}

// =====================================================================
// Kernel 3: per-(t,v) bitonic sort + score_mask + tail softmax + extra token
// =====================================================================
__global__ void __launch_bounds__(256) k_sort(const float* __restrict__ score,
                                              const float* __restrict__ img,
                                              int* __restrict__ keep_idx,
                                              float* __restrict__ extra,
                                              float* __restrict__ mask_out) {
  __shared__ float key[256];
  __shared__ int   sidx[256];
  __shared__ float nonw[NTAIL];
  __shared__ float sred[8];
  int b = blockIdx.x, v = b & 63;
  int tid = threadIdx.x, wid = tid >> 5, lane = tid & 31;

  key[tid]  = (tid < LV) ? score[(size_t)b * LV + tid] : -3.0e38f;
  sidx[tid] = (tid < LV) ? tid : 0xFFFF;

  for (int k = 2; k <= 256; k <<= 1) {
    for (int j = k >> 1; j > 0; j >>= 1) {
      __syncthreads();
      int i = tid, ixj = i ^ j;
      if (ixj > i) {
        bool desc = ((i & k) == 0);
        float ki = key[i], kj = key[ixj];
        int ii = sidx[i], ij = sidx[ixj];
        bool iFirst = (ki > kj) || (ki == kj && ii < ij);
        bool doswap = desc ? !iFirst : iFirst;
        if (doswap) { key[i] = kj; key[ixj] = ki; sidx[i] = ij; sidx[ixj] = ii; }
      }
    }
  }
  __syncthreads();

  { int l = sidx[tid];
    if (l < LV) mask_out[(size_t)b * LV + l] = (tid < NKEEP) ? 1.f : 0.f; }
  if (tid < NKEEP) keep_idx[(size_t)b * NKEEP + tid] = sidx[tid];

  float m = key[NKEEP];
  float e = (tid >= NKEEP && tid < LV) ? __expf(key[tid] - m) : 0.f;
  float p = lane_sum(e);
  if (lane == 0) sred[wid] = p;
  __syncthreads();
  float s = 0.f;
#pragma unroll
  for (int i = 0; i < 8; ++i) s += sred[i];
  if (tid >= NKEEP && tid < LV) nonw[tid - NKEEP] = e / s;
  __syncthreads();

  const float* ibase = img + (size_t)v * LV * CDIM;
  for (int c = tid; c < CDIM; c += 256) {
    float acc = 0.f;
    for (int j = 0; j < NTAIL; ++j)
      acc += nonw[j] * ibase[(size_t)sidx[NKEEP + j] * CDIM + c];
    extra[(size_t)b * CDIM + c] = acc;
  }
}

// =====================================================================
// Kernel 4: per-(t,v) heavy pipeline (LN, WMMA GEMMs, softmaxes, attention)
// =====================================================================
__global__ void __launch_bounds__(256) k_main(const float* __restrict__ img,
                                              const float* __restrict__ lgam,
                                              const float* __restrict__ lbet,
                                              const float* __restrict__ b1,
                                              const float* __restrict__ W2,
                                              const float* __restrict__ b2,
                                              const float* __restrict__ scale_p,
                                              const int* __restrict__ keep_idx,
                                              const float* __restrict__ extra,
                                              const unsigned* __restrict__ W1fm,
                                              const float* __restrict__ cap_nrm,
                                              float* __restrict__ out_sims) {
  extern __shared__ char smem[];
  unsigned* xnb_u   = (unsigned*)(smem + XNB_OFF);  // [112][260] dwords of bf16-pairs (xn)
  unsigned* tok_u   = (unsigned*)(smem + TOK_OFF);  // [64][516] packed token pairs / early: W1 frag-major
  unsigned* wab_u   = (unsigned*)(smem + WAB_OFF);  // [64][68]  packed softmax weights
  float*    hbuf    = (float*)(smem + H_OFF);       // [98][112] f32
  int*      kix     = (int*)(smem + KIX_OFF);
  float*    wsim    = (float*)(smem + WSIM_OFF);
  float*    wtmp    = (float*)smem;                 // alias xnb: [49][112] pre-softmax w
  float*    selnorm = (float*)smem;                 // alias xnb: [50][520] f32

  int b = blockIdx.x, t = b >> 6, v = b & 63;
  int tid = threadIdx.x, wid = tid >> 5, lane = tid & 31;

  if (tid < NKEEP) kix[tid] = keep_idx[(size_t)b * NKEEP + tid];

  // ---- Phase 0: async-stage fragment-major W1 into LDS (dense linear copy);
  //      overlaps with Phase 1 gather+LN. 114688 B / (256 thr * 16 B) = 28 iters.
  {
    const char* gbase = (const char*)W1fm;
    for (int it = 0; it < 28; ++it) {
      unsigned ofs = (unsigned)((it * 256 + tid) * 16);
      unsigned ldsoff = (unsigned)(TOK_OFF + ofs);
      const void* ga = gbase + ofs;
      asm volatile("global_load_async_to_lds_b128 %0, %1, off"
                   :: "v"(ldsoff), "v"(ga) : "memory");
    }
  }
  __syncthreads();   // kix visible

  // ---- Phase 1: gather kept tokens, LayerNorm -> xnb (pair-packed bf16) ----
  const float* ibase = img + (size_t)v * LV * CDIM;
  const float2* lg2 = (const float2*)lgam;
  const float2* lb2 = (const float2*)lbet;
  for (int j = wid; j < MP; j += 8) {
    unsigned* xrow = xnb_u + (size_t)j * XSTRW;
    if (j < NKEEP) {
      const float2* tp2 = (const float2*)(ibase + (size_t)kix[j] * CDIM);
      float2 xv[8]; float s = 0.f, s2 = 0.f;
#pragma unroll
      for (int i = 0; i < 8; ++i) {
        float2 x = tp2[lane + 32 * i];
        xv[i] = x;
        s += x.x + x.y; s2 += x.x * x.x + x.y * x.y;
      }
      s = lane_sum(s); s2 = lane_sum(s2);
      float mu = s * (1.f / CDIM);
      float var = s2 * (1.f / CDIM) - mu * mu;
      float rs = rsqrtf(var + LNEPS);
#pragma unroll
      for (int i = 0; i < 8; ++i) {
        int e = lane + 32 * i;
        float2 g = lg2[e], bb = lb2[e];
        float a0 = (xv[i].x - mu) * rs * g.x + bb.x;
        float a1 = (xv[i].y - mu) * rs * g.y + bb.y;
        xrow[e] = pack_bf16(a0, a1);
      }
    } else {
#pragma unroll
      for (int i = 0; i < 8; ++i) xrow[lane + 32 * i] = 0u;
    }
  }
  asm volatile("s_wait_asynccnt 0x0" ::: "memory");
  __syncthreads();

  // ---- Phase 2: GEMM1 h = gelu(xn @ W1 + b1) via WMMA bf16 ----
  // B fragments: 2x ds_load_b128 each (fragment-major LDS), no marshalling.
  if (wid < 7) {
    v8f acc[7];
#pragma unroll
    for (int q = 0; q < 7; ++q)
#pragma unroll
      for (int g = 0; g < 8; ++g) acc[q][g] = 0.f;
    int mrow0 = wid * 16;
    for (int kk = 0; kk < 16; ++kk) {
      v16bf a = load_a_frag_pk(xnb_u, XSTRW, mrow0, kk * 32, lane);
#pragma unroll
      for (int nt = 0; nt < 7; ++nt) {
        v16bf bb = load_b_frag_fm(tok_u, kk, nt, lane);
        acc[nt] = __builtin_amdgcn_wmma_f32_16x16x32_bf16(
            false, a, false, bb, (short)0, acc[nt], false, false);
      }
    }
    int mo = (lane >> 4) * 8, n0 = lane & 15;
#pragma unroll
    for (int nt = 0; nt < 7; ++nt) {
      int n = nt * 16 + n0;
#pragma unroll
      for (int g = 0; g < 8; ++g) {
        int m = mrow0 + mo + g;
        if (m < NKEEP) {
          float x = 0.f;
          if (n < HIDDEN) {
            x = acc[nt][g] + b1[n];
            x = 0.5f * x * (1.f + erff(x * 0.70710678f));
          }
          hbuf[m * HPAD + n] = x;
        }
      }
    }
  }
  __syncthreads();

  // ---- Phase 3: small GEMM2 wtmp[p][k] = scale*(h[k]@W2[:,p] + b2[p]) ----
  float sc = scale_p[0];
  for (int idx = tid; idx < NKEEP * KEEPED; idx += 256) {
    int k = idx / KEEPED, p = idx % KEEPED;
    float acc = b2[p];
    for (int hh = 0; hh < HIDDEN; ++hh)
      acc += hbuf[k * HPAD + hh] * W2[hh * KEEPED + p];
    wtmp[p * HPAD + k] = acc * sc;
  }
  __syncthreads();

  // ---- Phase 3b: softmax rows -> wab (pair-packed);
  //      plus re-gather raw tokens pair-packed -> tok_u (overwrites W1 staging) ----
  for (int p = wid; p < PPAD; p += 8) {
    unsigned* wrow = wab_u + (size_t)p * WABW;
    if (p < KEEPED) {
      int k10 = 2 * lane + 64, k11 = k10 + 1;
      float v00 = wtmp[p * HPAD + 2 * lane];
      float v01 = wtmp[p * HPAD + 2 * lane + 1];
      float v10 = (k10 < NKEEP) ? wtmp[p * HPAD + k10] : -1e30f;
      float v11 = (k11 < NKEEP) ? wtmp[p * HPAD + k11] : -1e30f;
      float mx = lane_max(fmaxf(fmaxf(v00, v01), fmaxf(v10, v11)));
      float e00 = __expf(v00 - mx), e01 = __expf(v01 - mx);
      float e10 = (k10 < NKEEP) ? __expf(v10 - mx) : 0.f;
      float e11 = (k11 < NKEEP) ? __expf(v11 - mx) : 0.f;
      float se = lane_sum(e00 + e01 + e10 + e11);
      float inv = 1.f / se;
      wrow[lane]      = pack_bf16(e00 * inv, e01 * inv);
      wrow[lane + 32] = pack_bf16(e10 * inv, e11 * inv);
    } else {
      wrow[lane] = 0u; wrow[lane + 32] = 0u;
    }
  }
  for (int jp = wid; jp < 64; jp += 8) {      // token rows (2jp, 2jp+1) -> one dword
    int j0 = 2 * jp, j1 = j0 + 1;
    unsigned* trow = tok_u + (size_t)jp * TOKW;
    if (j0 < NKEEP) {
      const float* r0 = ibase + (size_t)kix[j0] * CDIM;
      const float* r1 = (j1 < NKEEP) ? ibase + (size_t)kix[j1] * CDIM : r0;
      bool has1 = (j1 < NKEEP);
#pragma unroll
      for (int i = 0; i < 16; ++i) {
        int c = lane + 32 * i;
        float x0 = r0[c];
        float x1 = has1 ? r1[c] : 0.f;
        trow[c] = pack_bf16(x0, x1);
      }
    } else {
#pragma unroll
      for (int i = 0; i < 16; ++i) trow[lane + 32 * i] = 0u;
    }
  }
  __syncthreads();

  // ---- Phase 4: aggr = w @ tokens via WMMA (M=64, N=512, K=128) ----
  for (int ti = wid; ti < 128; ti += 8) {
    int mt = ti >> 5, nt = ti & 31;
    v8f acc;
#pragma unroll
    for (int g = 0; g < 8; ++g) acc[g] = 0.f;
#pragma unroll
    for (int kk = 0; kk < 4; ++kk) {
      v16bf a  = load_a_frag_pk(wab_u, WABW, mt * 16, kk * 32, lane);
      v16bf bb = load_b_frag_pk(tok_u, TOKW, kk * 32, nt * 16, lane);
      acc = __builtin_amdgcn_wmma_f32_16x16x32_bf16(
          false, a, false, bb, (short)0, acc, false, false);
    }
    int mo = (lane >> 4) * 8, n = nt * 16 + (lane & 15);
#pragma unroll
    for (int g = 0; g < 8; ++g) {
      int m = mt * 16 + mo + g;
      if (m < KEEPED) selnorm[m * SSN + n] = acc[g];
    }
  }
  for (int c = tid; c < CDIM; c += 256)
    selnorm[KEEPED * SSN + c] = extra[(size_t)b * CDIM + c];
  __syncthreads();

  // ---- l2norm the 50 rows in place ----
  for (int r = wid; r < KEEPED + 1; r += 8) {
    float ss = 0.f;
#pragma unroll
    for (int i = 0; i < 16; ++i) {
      int c = lane + 32 * i;
      float x = selnorm[r * SSN + c];
      ss += x * x;
    }
    ss = lane_sum(ss);
    float inv = 1.f / fmaxf(sqrtf(ss), 1e-12f);
#pragma unroll
    for (int i = 0; i < 16; ++i) {
      int c = lane + 32 * i;
      selnorm[r * SSN + c] *= inv;
    }
  }
  __syncthreads();

  // ---- Phase 5: per-word cross attention + cosine sim (3 words / wave) ----
  const float* capb = cap_nrm + (size_t)t * NWORD * CDIM;
  for (int w = wid; w < NWORD; w += 8) {
    const float* cw = capb + (size_t)w * CDIM;
    float d0 = 0.f, d1 = 0.f;
    { int p = lane;
      for (int c = 0; c < CDIM; ++c) d0 += cw[c] * selnorm[p * SSN + c]; }
    if (lane + 32 <= KEEPED) {
      int p = lane + 32;
      for (int c = 0; c < CDIM; ++c) d1 += cw[c] * selnorm[p * SSN + c];
    }
    float a0 = LAMBDA * d0;
    float a1 = (lane + 32 <= KEEPED) ? LAMBDA * d1 : -1e30f;
    float mx = lane_max(fmaxf(a0, a1));
    float e0 = __expf(a0 - mx);
    float e1 = (lane + 32 <= KEEPED) ? __expf(a1 - mx) : 0.f;
    float se = lane_sum(e0 + e1);
    float inv = 1.f / se;
    float att0 = e0 * inv, att1 = e1 * inv;

    float ctxc[16];
#pragma unroll
    for (int i = 0; i < 16; ++i) ctxc[i] = 0.f;
    for (int p = 0; p <= KEEPED; ++p) {
      float ap = (p < 32) ? __shfl(att0, p, 32) : __shfl(att1, p - 32, 32);
#pragma unroll
      for (int i = 0; i < 16; ++i)
        ctxc[i] += ap * selnorm[p * SSN + lane + 32 * i];
    }
    float ss = 0.f, dot = 0.f;
#pragma unroll
    for (int i = 0; i < 16; ++i) {
      float cc = cw[lane + 32 * i];
      ss += ctxc[i] * ctxc[i];
      dot += cc * ctxc[i];
    }
    ss = lane_sum(ss); dot = lane_sum(dot);
    if (lane == 0) wsim[w] = dot / fmaxf(sqrtf(ss), 1e-12f);
  }
  __syncthreads();
  if (tid == 0) {
    float s = 0.f;
    for (int w = 0; w < NWORD; ++w) s += wsim[w];
    out_sims[(size_t)v * BT + t] = s * (1.f / NWORD);  // improve_sims = sim.T
  }
}

// =====================================================================
extern "C" void kernel_launch(void* const* d_in, const int* in_sizes, int n_in,
                              void* d_out, int out_size, void* d_ws, size_t ws_size,
                              hipStream_t stream) {
  const float* img   = (const float*)d_in[0];
  const float* cap   = (const float*)d_in[1];
  const float* gamma = (const float*)d_in[3];
  const float* beta  = (const float*)d_in[4];
  const float* W1    = (const float*)d_in[5];
  const float* b1    = (const float*)d_in[6];
  const float* W2    = (const float*)d_in[7];
  const float* b2    = (const float*)d_in[8];
  const float* scale = (const float*)d_in[9];

  char* ws = (char*)d_ws;
  float* img_glo   = (float*)(ws + 0);                         //  64*512
  float* inv_img   = (float*)(ws + 131072);                    //  64*196
  float* self_attn = (float*)(ws + 181248);                    //  64*196
  float* cap_glo   = (float*)(ws + 231424);                    //  32*512
  float* cap_nrm   = (float*)(ws + 296960);                    //  32*24*512
  float* score     = (float*)(ws + 1869824);                   //  32*64*196
  int*   keep_idx  = (int*)  (ws + 3475456);                   //  32*64*98
  float* extra     = (float*)(ws + 4278272);                   //  32*64*512
  unsigned* W1fm   = (unsigned*)(ws + 8472576);                //  28672 dwords, fragment-major

  float* out       = (float*)d_out;
  float* out_sims  = out;               // (64,32)
  float* out_mask  = out + BV * BT;     // (32,64,196)

  hipFuncSetAttribute(reinterpret_cast<const void*>(k_main),
                      hipFuncAttributeMaxDynamicSharedMemorySize, SMEM_TOTAL);

  k_prep_img<<<BV, 256, 0, stream>>>(img, img_glo, inv_img, self_attn);
  k_prep_cap<<<BT, 256, 0, stream>>>(cap, cap_glo, cap_nrm);
  k_w1cvt<<<(W1FM_DWORDS + 255) / 256, 256, 0, stream>>>(W1, W1fm);
  k_score<<<dim3(LV, BV), 256, 0, stream>>>(img, inv_img, self_attn, cap_glo, score);
  k_sort<<<BT * BV, 256, 0, stream>>>(score, img, keep_idx, extra, out_mask);
  k_main<<<BT * BV, 256, SMEM_TOTAL, stream>>>(img, gamma, beta, b1, W2, b2, scale,
                                               keep_idx, extra, W1fm, cap_nrm, out_sims);
}
